// Seq2Seq_36429912604992
// MI455X (gfx1250) — matrix operand order repaired and checked by reading, not compiled, not run on hardware
//
#include <hip/hip_runtime.h>
#include <hip/hip_bf16.h>

// Problem constants (fixed by the reference)
#define BB 32
#define SS 128
#define TT 128
#define EE 256
#define HH 512
#define H3 1536
#define VT 16000

typedef __attribute__((ext_vector_type(16))) __bf16 v16bf;
typedef __attribute__((ext_vector_type(8)))  __bf16 v8bf;
typedef __attribute__((ext_vector_type(8)))  float  v8f;

// ---------------------------------------------------------------------------
// bf16 WMMA GEMM, 2x2 register-blocked: each wave computes a 32x32 C block.
// C[M,N] = A[M,K](bf16) @ W[N,K](bf16)^T (+bias), f32 accumulate/output.
// grid = (N/32, M/32), block = 32 (one wave; EXEC all-ones for WMMA).
// flags bit0: output swizzle m=(t*B+b) -> C[b][t][n] with B=32, T=128.
// ---------------------------------------------------------------------------
__device__ __forceinline__ v16bf load_a_frag(const __bf16* p) {
    // A fragment 16x32 (MxK), 16-bit layout: per lane two 8-elem chunks,
    // chunk0 at k0+8*half, chunk1 at +16 (half baked into p by caller).
    v8bf lo = *(const v8bf*)(p);
    v8bf hi = *(const v8bf*)(p + 16);
    return __builtin_shufflevector(lo, hi, 0,1,2,3,4,5,6,7,8,9,10,11,12,13,14,15);
}
__device__ __forceinline__ v16bf load_b_frag(const __bf16* p) {
    // B fragment 32x16 (KxN): per lane two 8-elem chunks,
    // chunk0 at k0+8*half, chunk1 at +16 (half baked into p by caller).
    v8bf lo = *(const v8bf*)(p);
    v8bf hi = *(const v8bf*)(p + 16);
    return __builtin_shufflevector(lo, hi, 0,1,2,3,4,5,6,7,8,9,10,11,12,13,14,15);
}

__device__ __forceinline__ void store_tile(float* __restrict__ C, int ldc,
                                           int m16, int col, int half,
                                           const v8f& acc, float bv, int flags) {
    // C/D layout: lanes 0-15 -> M = r, lanes 16-31 -> M = 8 + r; N = lane&15
    #pragma unroll
    for (int r = 0; r < 8; ++r) {
        const int m = m16 + 8 * half + r;
        size_t off;
        if (flags & 1) {
            const int b_ = m & (BB - 1);
            const int t_ = m >> 5;
            off = ((size_t)b_ * TT + t_) * (size_t)ldc + col;
        } else {
            off = (size_t)m * ldc + col;
        }
        C[off] = acc[r] + bv;
    }
}

__global__ void __launch_bounds__(32) wmma_gemm_k(
    const __bf16* __restrict__ A, int lda,
    const __bf16* __restrict__ W, int ldw,
    const float* __restrict__ bias,
    float* __restrict__ C, int ldc,
    int K, int flags)
{
    const int lane = threadIdx.x;
    const int idx  = lane & 15;
    const int half = lane >> 4;
    const int n32  = blockIdx.x * 32;
    const int m32  = blockIdx.y * 32;

    const __bf16* __restrict__ a0 = A + (size_t)(m32 + idx) * lda + 8 * half;
    const __bf16* __restrict__ a1 = a0 + (size_t)16 * lda;
    const __bf16* __restrict__ w0 = W + (size_t)(n32 + idx) * ldw + 8 * half;
    const __bf16* __restrict__ w1 = w0 + (size_t)16 * ldw;

    v8f acc00 = {}, acc01 = {}, acc10 = {}, acc11 = {};
    for (int k0 = 0; k0 < K; k0 += 32) {
        const v16bf af0 = load_a_frag(a0 + k0);
        const v16bf af1 = load_a_frag(a1 + k0);
        const v16bf bf0 = load_b_frag(w0 + k0);
        const v16bf bf1 = load_b_frag(w1 + k0);
        acc00 = __builtin_amdgcn_wmma_f32_16x16x32_bf16(false, af0, false, bf0, (short)0, acc00, false, false);
        acc01 = __builtin_amdgcn_wmma_f32_16x16x32_bf16(false, af0, false, bf1, (short)0, acc01, false, false);
        acc10 = __builtin_amdgcn_wmma_f32_16x16x32_bf16(false, af1, false, bf0, (short)0, acc10, false, false);
        acc11 = __builtin_amdgcn_wmma_f32_16x16x32_bf16(false, af1, false, bf1, (short)0, acc11, false, false);
    }

    const int   col0 = n32 + idx;
    const int   col1 = col0 + 16;
    const float bv0  = bias ? bias[col0] : 0.0f;
    const float bv1  = bias ? bias[col1] : 0.0f;
    store_tile(C, ldc, m32,      col0, half, acc00, bv0, flags);
    store_tile(C, ldc, m32,      col1, half, acc01, bv1, flags);
    store_tile(C, ldc, m32 + 16, col0, half, acc10, bv0, flags);
    store_tile(C, ldc, m32 + 16, col1, half, acc11, bv1, flags);
}

// ---------------------------------------------------------------------------
// Pack / gather / elementwise / reduction kernels
// ---------------------------------------------------------------------------
__global__ void pack_bf16_k(const float* __restrict__ in, __bf16* __restrict__ out, int n) {
    int i = blockIdx.x * blockDim.x + threadIdx.x;
    if (i < n) out[i] = (__bf16)in[i];
}

__global__ void zero_f32_k(float* p, int n) {
    int i = blockIdx.x * blockDim.x + threadIdx.x;
    if (i < n) p[i] = 0.0f;
}
__global__ void zero_bf16_k(__bf16* p, int n) {
    int i = blockIdx.x * blockDim.x + threadIdx.x;
    if (i < n) p[i] = (__bf16)0.0f;
}

// out time-major bf16: out[(s*B + b)*E + e] = bf16(emb[tok[b*S + s]][e])
__global__ void embed_gather_k(const int* __restrict__ tok,
                               const float* __restrict__ emb,
                               __bf16* __restrict__ out, int Bn, int Sn, int En) {
    int i = blockIdx.x * blockDim.x + threadIdx.x;
    if (i >= Bn * Sn * En) return;
    int e  = i % En;
    int sb = i / En;
    int b  = sb % Bn;
    int s  = sb / Bn;
    int t  = tok[b * Sn + s];
    out[i] = (__bf16)emb[(size_t)t * En + e];
}

// GRU gate fuse: gi = gi1 (+ gi2); writes f32 h (recurrence) + bf16 shadow (GEMM A).
__global__ void gru_cell_k(const float* __restrict__ gi1,
                           const float* __restrict__ gi2,
                           const float* __restrict__ gh,
                           const float* __restrict__ h_in,
                           float* __restrict__ h_out,
                           __bf16* __restrict__ h_out_bf, int Bn, int Hn) {
    int i = blockIdx.x * blockDim.x + threadIdx.x;
    if (i >= Bn * Hn) return;
    int b = i / Hn, j = i % Hn;
    const float* g1 = gi1 + (size_t)b * 3 * Hn;
    float ir = g1[j], iz = g1[j + Hn], in_ = g1[j + 2 * Hn];
    if (gi2) {
        const float* g2 = gi2 + (size_t)b * 3 * Hn;
        ir += g2[j]; iz += g2[j + Hn]; in_ += g2[j + 2 * Hn];
    }
    const float* gH = gh + (size_t)b * 3 * Hn;
    float r = 1.0f / (1.0f + __expf(-(ir + gH[j])));
    float z = 1.0f / (1.0f + __expf(-(iz + gH[j + Hn])));
    float n = tanhf(in_ + r * gH[j + 2 * Hn]);
    float h = (1.0f - z) * n + z * h_in[i];
    h_out[i]    = h;
    h_out_bf[i] = (__bf16)h;
}

// scores[b*S + s] = sum_h tanh(hW[b,h] + enc_proj[s,b,h]) * v[h]; one wave per (s,b)
__global__ void __launch_bounds__(32) attn_scores_k(
    const float* __restrict__ hW, const float* __restrict__ enc_proj,
    const float* __restrict__ v, float* __restrict__ scores, int Bn, int Sn, int Hn) {
    int sb = blockIdx.x;
    int b = sb % Bn, s = sb / Bn;
    int lane = threadIdx.x;
    const float* ep = enc_proj + (size_t)sb * Hn;
    const float* hw = hW + (size_t)b * Hn;
    float acc = 0.0f;
    for (int h = lane; h < Hn; h += 32)
        acc += tanhf(hw[h] + ep[h]) * v[h];
    #pragma unroll
    for (int off = 16; off > 0; off >>= 1)
        acc += __shfl_xor(acc, off, 32);
    if (lane == 0) scores[b * Sn + s] = acc;
}

// softmax over s; one block (Sn threads) per b
__global__ void softmax_k(const float* __restrict__ scores,
                          float* __restrict__ w, int Sn) {
    __shared__ float red[SS];
    int b = blockIdx.x, s = threadIdx.x;
    float v = scores[b * Sn + s];
    red[s] = v; __syncthreads();
    for (int off = Sn >> 1; off > 0; off >>= 1) {
        if (s < off) red[s] = fmaxf(red[s], red[s + off]);
        __syncthreads();
    }
    float mx = red[0]; __syncthreads();
    float e = __expf(v - mx);
    red[s] = e; __syncthreads();
    for (int off = Sn >> 1; off > 0; off >>= 1) {
        if (s < off) red[s] += red[s + off];
        __syncthreads();
    }
    w[b * Sn + s] = e / red[0];
}

// ctx[b,h] = sum_s w[b,s] * enc_seq[s,b,h]; bf16 output (feeds GEMM only)
__global__ void attn_ctx_k(const float* __restrict__ w,
                           const float* __restrict__ enc_seq,
                           __bf16* __restrict__ ctx, int Bn, int Sn, int Hn) {
    int i = blockIdx.x * blockDim.x + threadIdx.x;
    if (i >= Bn * Hn) return;
    int b = i / Hn, h = i % Hn;
    float acc = 0.0f;
    for (int s = 0; s < Sn; ++s)
        acc += w[b * Sn + s] * enc_seq[((size_t)s * Bn + b) * Hn + h];
    ctx[i] = (__bf16)acc;
}

// ---------------------------------------------------------------------------
// Host orchestration
// ---------------------------------------------------------------------------
extern "C" void kernel_launch(void* const* d_in, const int* in_sizes, int n_in,
                              void* d_out, int out_size, void* d_ws, size_t ws_size,
                              hipStream_t stream) {
    (void)in_sizes; (void)n_in; (void)out_size; (void)ws_size;
    const int*   src     = (const int*)  d_in[0];
    const int*   tgt     = (const int*)  d_in[1];
    const float* emb_src = (const float*)d_in[2];
    const float* emb_tgt = (const float*)d_in[3];
    const float* enc_Wih = (const float*)d_in[4];
    const float* enc_Whh = (const float*)d_in[5];
    const float* enc_bih = (const float*)d_in[6];
    const float* enc_bhh = (const float*)d_in[7];
    const float* dec_Wih = (const float*)d_in[8];
    const float* dec_Whh = (const float*)d_in[9];
    const float* dec_bih = (const float*)d_in[10];
    const float* dec_bhh = (const float*)d_in[11];
    const float* attn_W  = (const float*)d_in[12];
    const float* attn_b  = (const float*)d_in[13];
    const float* attn_v  = (const float*)d_in[14];
    const float* fc_W    = (const float*)d_in[15];
    const float* fc_b    = (const float*)d_in[16];
    float* out = (float*)d_out;

    // ---- Workspace carve-up: bf16 region first (32B-aligned chunks), f32 after
    __bf16* q = (__bf16*)d_ws;
    __bf16* xs_bf       = q; q += (size_t)SS * BB * EE;   // (S,B,E)
    __bf16* ys_bf       = q; q += (size_t)TT * BB * EE;   // (T,B,E)
    __bf16* enc_seq_bf  = q; q += (size_t)SS * BB * HH;   // (S,B,H)
    __bf16* dec_seq_bf  = q; q += (size_t)TT * BB * HH;   // (T,B,H)
    __bf16* h0_bf       = q; q += (size_t)BB * HH;
    __bf16* ctx_bf      = q; q += (size_t)BB * HH;
    __bf16* enc_Wih_bf  = q; q += (size_t)H3 * EE;
    __bf16* enc_Whh_bf  = q; q += (size_t)H3 * HH;
    __bf16* dec_Wih_bf  = q; q += (size_t)H3 * (EE + HH);
    __bf16* dec_Whh_bf  = q; q += (size_t)H3 * HH;
    __bf16* attn_W_bf   = q; q += (size_t)HH * 2 * HH;
    __bf16* fc_W_bf     = q; q += (size_t)VT * HH;

    float* p = (float*)q;
    float* Gi       = p; p += (size_t)SS * BB * H3;   // xs @ enc_Wih^T + bih
    float* Ypre     = p; p += (size_t)TT * BB * H3;   // ys @ dec_Wih[:, :E]^T
    float* enc_seq  = p; p += (size_t)SS * BB * HH;   // f32 (for ctx + recurrence)
    float* enc_proj = p; p += (size_t)SS * BB * HH;
    float* h0       = p; p += (size_t)BB * HH;
    float* hbufA    = p; p += (size_t)BB * HH;        // decoder h ping-pong
    float* hbufB    = p; p += (size_t)BB * HH;
    float* ghbuf    = p; p += (size_t)BB * H3;
    float* giC      = p; p += (size_t)BB * H3;
    float* hW       = p; p += (size_t)BB * HH;
    float* scores   = p; p += (size_t)BB * SS;
    float* wts      = p; p += (size_t)BB * SS;

    const int MB = SS * BB;   // 4096 rows in the big batched GEMMs
    #define CEIL256(n) (((n) + 255) / 256)

    // ---- One-time packing of weights to bf16
    pack_bf16_k<<<CEIL256(H3 * EE),        256, 0, stream>>>(enc_Wih, enc_Wih_bf, H3 * EE);
    pack_bf16_k<<<CEIL256(H3 * HH),        256, 0, stream>>>(enc_Whh, enc_Whh_bf, H3 * HH);
    pack_bf16_k<<<CEIL256(H3 * (EE + HH)), 256, 0, stream>>>(dec_Wih, dec_Wih_bf, H3 * (EE + HH));
    pack_bf16_k<<<CEIL256(H3 * HH),        256, 0, stream>>>(dec_Whh, dec_Whh_bf, H3 * HH);
    pack_bf16_k<<<CEIL256(HH * 2 * HH),    256, 0, stream>>>(attn_W, attn_W_bf, HH * 2 * HH);
    pack_bf16_k<<<CEIL256(VT * HH),        256, 0, stream>>>(fc_W, fc_W_bf, VT * HH);

    // ---- h0 = 0 (f32 + bf16 shadow)
    zero_f32_k <<<CEIL256(BB * HH), 256, 0, stream>>>(h0, BB * HH);
    zero_bf16_k<<<CEIL256(BB * HH), 256, 0, stream>>>(h0_bf, BB * HH);

    // ---- Embedding gathers (time-major, straight to bf16)
    embed_gather_k<<<CEIL256(SS * BB * EE), 256, 0, stream>>>(src, emb_src, xs_bf, BB, SS, EE);
    embed_gather_k<<<CEIL256(TT * BB * EE), 256, 0, stream>>>(tgt, emb_tgt, ys_bf, BB, TT, EE);

    // ---- Hoisted input projections across all timesteps
    wmma_gemm_k<<<dim3(H3 / 32, MB / 32), 32, 0, stream>>>(
        xs_bf, EE, enc_Wih_bf, EE, enc_bih, Gi, H3, EE, 0);
    wmma_gemm_k<<<dim3(H3 / 32, MB / 32), 32, 0, stream>>>(
        ys_bf, EE, dec_Wih_bf, EE + HH, nullptr, Ypre, H3, EE, 0);

    // ---- Encoder scan
    const float*  hprev    = h0;
    const __bf16* hprev_bf = h0_bf;
    for (int s = 0; s < SS; ++s) {
        wmma_gemm_k<<<dim3(H3 / 32, 1), 32, 0, stream>>>(
            hprev_bf, HH, enc_Whh_bf, HH, enc_bhh, ghbuf, H3, HH, 0);
        float*  hout    = enc_seq    + (size_t)s * BB * HH;
        __bf16* hout_bf = enc_seq_bf + (size_t)s * BB * HH;
        gru_cell_k<<<CEIL256(BB * HH), 256, 0, stream>>>(
            Gi + (size_t)s * BB * H3, nullptr, ghbuf, hprev, hout, hout_bf, BB, HH);
        hprev = hout; hprev_bf = hout_bf;
    }

    // ---- enc_proj = enc_seq @ We^T + attn_b  (We = attn_W[:, H:], ldw = 2H)
    wmma_gemm_k<<<dim3(HH / 32, MB / 32), 32, 0, stream>>>(
        enc_seq_bf, HH, attn_W_bf + HH, 2 * HH, attn_b, enc_proj, HH, HH, 0);

    // ---- Decoder scan
    for (int t = 0; t < TT; ++t) {
        wmma_gemm_k<<<dim3(HH / 32, 1), 32, 0, stream>>>(
            hprev_bf, HH, attn_W_bf, 2 * HH, nullptr, hW, HH, HH, 0);
        attn_scores_k<<<SS * BB, 32, 0, stream>>>(hW, enc_proj, attn_v, scores, BB, SS, HH);
        softmax_k<<<BB, SS, 0, stream>>>(scores, wts, SS);
        attn_ctx_k<<<CEIL256(BB * HH), 256, 0, stream>>>(wts, enc_seq, ctx_bf, BB, SS, HH);
        wmma_gemm_k<<<dim3(H3 / 32, 1), 32, 0, stream>>>(
            ctx_bf, HH, dec_Wih_bf + EE, EE + HH, dec_bih, giC, H3, HH, 0);
        wmma_gemm_k<<<dim3(H3 / 32, 1), 32, 0, stream>>>(
            hprev_bf, HH, dec_Whh_bf, HH, dec_bhh, ghbuf, H3, HH, 0);
        float*  hout    = (t & 1) ? hbufB : hbufA;
        __bf16* hout_bf = dec_seq_bf + (size_t)t * BB * HH;
        gru_cell_k<<<CEIL256(BB * HH), 256, 0, stream>>>(
            giC, Ypre + (size_t)t * BB * H3, ghbuf, hprev, hout, hout_bf, BB, HH);
        hprev = hout; hprev_bf = hout_bf;
    }

    // ---- Final logits: dec_seq (4096 x 512) @ fc_W^T (16000 x 512) + fc_b,
    //      written swizzled to out[b][t][v]  (flags = 1)
    wmma_gemm_k<<<dim3(VT / 32, (TT * BB) / 32), 32, 0, stream>>>(
        dec_seq_bf, HH, fc_W_bf, HH, fc_b, out, VT, HH, 1);
    #undef CEIL256
}
